// MGCN_20486994002070
// MI455X (gfx1250) — compile-verified, matrix operand order checked by reading, not run on hardware
//
#include <hip/hip_runtime.h>

// ---------------------------------------------------------------------------
// MGCN forward for MI455X (gfx1250, wave32, WMMA).
// All dense GEMMs run on v_wmma_f32_16x16x32_bf16 (f32 accumulate).
// Roofline: ~0.3 TFLOP, ~4 GB HBM traffic -> memory-bound at 23.3 TB/s when
// the matrix pipes are used; bf16 activations halve traffic vs f32.
// ---------------------------------------------------------------------------

#define N_NODES 40000
#define N_EDGES 400000
#define FDIM    128
#define NLAYERS 3
#define RDIM    510
#define RPAD    512
#define CUTOFF_F 51.0f

#define CDIV(a, b) (((a) + (b) - 1) / (b))

typedef __attribute__((ext_vector_type(8)))  __bf16 v8bf;
typedef __attribute__((ext_vector_type(16))) __bf16 v16bf;
typedef __attribute__((ext_vector_type(8)))  float  v8f;

// Load one wave's A (or B) fragment for a 16x32 bf16 tile, row-major source.
// Per ISA 16-bit A layout: lane r (0..15) = row, lane group (hi) selects
// K = k0+hi*8..+7 in elements 0..7 and K = k0+16+hi*8..+7 in elements 8..15.
// Caller bakes (row*ld + hi*8) into p; this reads p[0..7] and p[16..23].
__device__ __forceinline__ v16bf ld_frag16(const __bf16* __restrict__ p) {
  v8bf lo = *(const v8bf*)(p);
  v8bf hi = *(const v8bf*)(p + 16);
  v16bf r;
#pragma unroll
  for (int i = 0; i < 8; ++i) { r[i] = lo[i]; r[i + 8] = hi[i]; }
  return r;
}

// ACT: 0 = none, 1 = scaled softplus  sp(x) = x>28 ? x : 2*log1p(exp(x/2)),
//      2 = PReLU with scalar alpha.
template<int ACT>
__device__ __forceinline__ float actf(float v, float a) {
  if (ACT == 1) return (v > 28.0f) ? v : 2.0f * __logf(1.0f + __expf(0.5f * v));
  if (ACT == 2) return (v >= 0.0f) ? v : a * v;
  return v;
}

// ---------------------------------------------------------------------------
// Fused WMMA GEMM:  out = epilogue( A[M,K]bf16 @ Bt[128,K]bf16^T + bias[128] )
//   Bt is the pre-transposed weight (W^T, row-major [N=128, K]).
//   Optional f32 residual add, optional f32 store, optional bf16 store with
//   arbitrary row stride (used to write slices of the [N,512] concat buffer).
// Block = 256 threads = 8 waves; block tile 32(M) x 128(N);
// wave (mt, nt) owns a 16x32 tile => two 16x16 f32 accumulators.
// No divergence before the WMMAs (EXEC all ones, as the ISA requires).
// ---------------------------------------------------------------------------
template<int ACT>
__global__ __launch_bounds__(256)
void k_gemm_bf16(const __bf16* __restrict__ A, int lda,
                 const __bf16* __restrict__ Bt,
                 const float*  __restrict__ bias,
                 const float*  __restrict__ resid,   // may be null, ld = 128
                 float*        __restrict__ outF,    // may be null, ld = 128
                 __bf16*       __restrict__ outB,    // may be null
                 int ldob,
                 const float*  __restrict__ pa, int pai,
                 int M, int K)
{
  const int lane = threadIdx.x & 31;
  const int wave = threadIdx.x >> 5;
  const int r    = lane & 15;       // row (A) / column (B) within tile
  const int hi   = lane >> 4;       // lane-group -> K phase
  const int m0   = blockIdx.x * 32 + (wave >> 2) * 16;
  const int n0   = (wave & 3) * 32;

  int rowA = m0 + r; if (rowA >= M) rowA = M - 1;   // clamp (stores guarded)

  const __bf16* aP  = A  + (size_t)rowA * lda + hi * 8;
  const __bf16* b0P = Bt + (size_t)(n0      + r) * K + hi * 8;
  const __bf16* b1P = Bt + (size_t)(n0 + 16 + r) * K + hi * 8;

  v8f acc0 = {};
  v8f acc1 = {};

  for (int k0 = 0; k0 < K; k0 += 32) {
    v16bf a  = ld_frag16(aP  + k0);
    v16bf b0 = ld_frag16(b0P + k0);
    v16bf b1 = ld_frag16(b1P + k0);
    acc0 = __builtin_amdgcn_wmma_f32_16x16x32_bf16(
        false, a, false, b0, (short)0, acc0, false, false);
    acc1 = __builtin_amdgcn_wmma_f32_16x16x32_bf16(
        false, a, false, b1, (short)0, acc1, false, false);
  }

  const float aval = (ACT == 2) ? pa[pai] : 0.0f;
  const float bia0 = bias[n0 + r];
  const float bia1 = bias[n0 + 16 + r];

  // C/D layout: element i -> row m0 + hi*8 + i, col n0 + r (ISA 16x16 f32 C/D).
#pragma unroll
  for (int i = 0; i < 8; ++i) {
    int row = m0 + hi * 8 + i;
    if (row >= M) continue;
    {
      int col = n0 + r;
      float v = actf<ACT>(acc0[i] + bia0, aval);
      if (resid) v += resid[(size_t)row * FDIM + col];
      if (outF)  outF[(size_t)row * FDIM + col] = v;
      if (outB)  outB[(size_t)row * ldob + col] = (__bf16)v;
    }
    {
      int col = n0 + 16 + r;
      float v = actf<ACT>(acc1[i] + bia1, aval);
      if (resid) v += resid[(size_t)row * FDIM + col];
      if (outF)  outF[(size_t)row * FDIM + col] = v;
      if (outB)  outB[(size_t)row * ldob + col] = (__bf16)v;
    }
  }
}

// --- Weight pack: Bt[n*Kp + k] = bf16( W[k*128 + n] ), zero-pad k >= K0. ----
__global__ void k_pack_wt(const float* __restrict__ W, __bf16* __restrict__ Bt,
                          int K0, int Kp) {
  int t = blockIdx.x * blockDim.x + threadIdx.x;
  if (t >= FDIM * Kp) return;
  int n = t / Kp, k = t % Kp;
  Bt[t] = (__bf16)((k < K0) ? W[(size_t)k * FDIM + n] : 0.0f);
}

// --- RBF: rbf[e, k] = exp(-10*(|efeats[e]| - c_k)^2), bf16, K padded 512. ---
__global__ void k_rbf(const float* __restrict__ ef, __bf16* __restrict__ rbf) {
  int e = blockIdx.x;
  float x = ef[e * 3 + 0], y = ef[e * 3 + 1], z = ef[e * 3 + 2];
  float dist = sqrtf(x * x + y * y + z * z);
  int base = threadIdx.x * 4;
#pragma unroll
  for (int j = 0; j < 4; ++j) {
    int k = base + j;
    float v = 0.0f;
    if (k < RDIM) {
      float c  = CUTOFF_F * (float)k / (float)(RDIM - 1);
      float dx = dist - c;
      v = __expf(-10.0f * dx * dx);
    }
    rbf[(size_t)e * RPAD + k] = (__bf16)v;
  }
}

// --- Node embedding init: h f32 + bf16 copy into concat slice 0. ------------
__global__ void k_node_init(const int* __restrict__ nf,
                            const float* __restrict__ node_emb,
                            float* __restrict__ h, __bf16* __restrict__ xcat) {
  int t = blockIdx.x * blockDim.x + threadIdx.x;
  if (t >= N_NODES * FDIM) return;
  int n = t >> 7, f = t & 127;
  float v = node_emb[nf[n] * FDIM + f];
  h[t] = v;
  xcat[(size_t)n * (FDIM * (NLAYERS + 1)) + f] = (__bf16)v;
}

// --- Edge embedding init: e_bf = bf16(edge_emb[nt_s*nt_d + nt_s + nt_d]). ---
__global__ void k_edge_init(const int* __restrict__ nf,
                            const int* __restrict__ src,
                            const int* __restrict__ dst,
                            const float* __restrict__ edge_emb,
                            __bf16* __restrict__ ebf) {
  int t = blockIdx.x * blockDim.x + threadIdx.x;
  if (t >= N_EDGES * FDIM) return;
  int e = t >> 7, f = t & 127;
  int a = nf[src[e]], b = nf[dst[e]];
  int et = a * b + a + b;
  ebf[t] = (__bf16)edge_emb[et * FDIM + f];
}

__global__ void k_zero_f32(float* __restrict__ p, int cnt) {
  int t = blockIdx.x * blockDim.x + threadIdx.x;
  if (t < cnt) p[t] = 0.0f;
}

__global__ void k_f32_to_bf16(const float* __restrict__ s,
                              __bf16* __restrict__ d, int cnt) {
  int t = blockIdx.x * blockDim.x + threadIdx.x;
  if (t < cnt) d[t] = (__bf16)s[t];
}

// --- msg = hn[src]*d + e2 ; agg[dst] += msg  (4 features / thread). ---------
__global__ void k_msg_agg(const __bf16* __restrict__ hn,
                          const float*  __restrict__ dmat,
                          const __bf16* __restrict__ e2,
                          const int* __restrict__ src,
                          const int* __restrict__ dst,
                          float* __restrict__ agg) {
  int t = blockIdx.x * blockDim.x + threadIdx.x;
  if (t >= N_EDGES * 32) return;
  int e  = t >> 5;
  int f4 = (t & 31) * 4;
  int s  = src[e], dn = dst[e];
#pragma unroll
  for (int j = 0; j < 4; ++j) {
    int f = f4 + j;
    float m = (float)hn[(size_t)s * FDIM + f] * dmat[(size_t)e * FDIM + f]
            + (float)e2[(size_t)e * FDIM + f];
    atomicAdd(&agg[(size_t)dn * FDIM + f], m);
  }
}

// --- Final projection [128] -> [1]. -----------------------------------------
__global__ void k_dec4(const __bf16* __restrict__ x,
                       const float* __restrict__ w,
                       const float* __restrict__ b,
                       float* __restrict__ out) {
  int n = blockIdx.x * blockDim.x + threadIdx.x;
  if (n >= N_NODES) return;
  float acc = b[0];
#pragma unroll 8
  for (int f = 0; f < FDIM; ++f)
    acc += (float)x[(size_t)n * FDIM + f] * w[f];
  out[n] = acc;
}

// ---------------------------------------------------------------------------
extern "C" void kernel_launch(void* const* d_in, const int* in_sizes, int n_in,
                              void* d_out, int out_size, void* d_ws, size_t ws_size,
                              hipStream_t stream) {
  const int*   nfeats   = (const int*)  d_in[0];
  const float* efeats   = (const float*)d_in[1];
  const int*   src      = (const int*)  d_in[2];
  const int*   dst      = (const int*)  d_in[3];
  const float* node_emb = (const float*)d_in[4];
  const float* edge_emb = (const float*)d_in[5];
  const float* nl1_W = (const float*)d_in[6];  const float* nl1_b = (const float*)d_in[7];
  const float* d1_W  = (const float*)d_in[8];  const float* d1_b  = (const float*)d_in[9];
  const float* d2_W  = (const float*)d_in[10]; const float* d2_b  = (const float*)d_in[11];
  const float* eu_W  = (const float*)d_in[12]; const float* eu_b  = (const float*)d_in[13];
  const float* nl2_W = (const float*)d_in[14]; const float* nl2_b = (const float*)d_in[15];
  const float* nl3_W = (const float*)d_in[16]; const float* nl3_b = (const float*)d_in[17];
  const float* el1_W = (const float*)d_in[18]; const float* el1_b = (const float*)d_in[19];
  const float* dec0_W = (const float*)d_in[20]; const float* dec0_b = (const float*)d_in[21];
  const float* dec1_W = (const float*)d_in[22]; const float* dec1_b = (const float*)d_in[23];
  const float* dec2_W = (const float*)d_in[24]; const float* dec2_b = (const float*)d_in[25];
  const float* dec3_W = (const float*)d_in[26]; const float* dec3_b = (const float*)d_in[27];
  const float* dec4_W = (const float*)d_in[28]; const float* dec4_b = (const float*)d_in[29];
  const float* prelu_a = (const float*)d_in[30];
  float* out = (float*)d_out;

  // Bump allocator over d_ws (~1.05 GB total).
  char* wsb = (char*)d_ws;
  size_t off = 0;
  auto alloc = [&](size_t bytes) -> void* {
    off = (off + 255) & ~(size_t)255;
    void* p = wsb + off;
    off += bytes;
    return p;
  };

  const int XC = FDIM * (NLAYERS + 1);  // 512, concat row stride

  __bf16* rbf   = (__bf16*)alloc((size_t)N_EDGES * RPAD * 2);   // 409.6 MB
  float*  h     = (float*) alloc((size_t)N_NODES * FDIM * 4);
  __bf16* xcat  = (__bf16*)alloc((size_t)N_NODES * XC * 2);
  __bf16* hn_bf = (__bf16*)alloc((size_t)N_NODES * FDIM * 2);
  float*  agg   = (float*) alloc((size_t)N_NODES * FDIM * 4);
  __bf16* aggbf = (__bf16*)alloc((size_t)N_NODES * FDIM * 2);
  __bf16* t_bf  = (__bf16*)alloc((size_t)N_NODES * FDIM * 2);
  __bf16* e_bf  = (__bf16*)alloc((size_t)N_EDGES * FDIM * 2);
  __bf16* e2_bf = (__bf16*)alloc((size_t)N_EDGES * FDIM * 2);
  __bf16* dp_bf = (__bf16*)alloc((size_t)N_EDGES * FDIM * 2);
  float*  d_f32 = (float*) alloc((size_t)N_EDGES * FDIM * 4);   // 204.8 MB
  __bf16* decA  = (__bf16*)alloc((size_t)N_NODES * FDIM * 2);
  __bf16* decB  = (__bf16*)alloc((size_t)N_NODES * FDIM * 2);

  __bf16 *nl1t[NLAYERS], *d1t[NLAYERS], *d2t[NLAYERS], *eut[NLAYERS],
         *nl2t[NLAYERS], *nl3t[NLAYERS], *el1t[NLAYERS];
  for (int l = 0; l < NLAYERS; ++l) {
    nl1t[l] = (__bf16*)alloc((size_t)FDIM * FDIM * 2);
    d1t[l]  = (__bf16*)alloc((size_t)FDIM * RPAD * 2);
    d2t[l]  = (__bf16*)alloc((size_t)FDIM * FDIM * 2);
    eut[l]  = (__bf16*)alloc((size_t)FDIM * FDIM * 2);
    nl2t[l] = (__bf16*)alloc((size_t)FDIM * FDIM * 2);
    nl3t[l] = (__bf16*)alloc((size_t)FDIM * FDIM * 2);
    el1t[l] = (__bf16*)alloc((size_t)FDIM * FDIM * 2);
  }
  __bf16* dec0t = (__bf16*)alloc((size_t)FDIM * XC * 2);
  __bf16* dec1t = (__bf16*)alloc((size_t)FDIM * FDIM * 2);
  __bf16* dec2t = (__bf16*)alloc((size_t)FDIM * FDIM * 2);
  __bf16* dec3t = (__bf16*)alloc((size_t)FDIM * FDIM * 2);

  auto pack = [&](const float* W, __bf16* Bt, int K0, int Kp) {
    int cnt = FDIM * Kp;
    k_pack_wt<<<CDIV(cnt, 256), 256, 0, stream>>>(W, Bt, K0, Kp);
  };
  for (int l = 0; l < NLAYERS; ++l) {
    pack(nl1_W + (size_t)l * FDIM * FDIM, nl1t[l], FDIM, FDIM);
    pack(d1_W  + (size_t)l * RDIM * FDIM, d1t[l],  RDIM, RPAD);
    pack(d2_W  + (size_t)l * FDIM * FDIM, d2t[l],  FDIM, FDIM);
    pack(eu_W  + (size_t)l * FDIM * FDIM, eut[l],  FDIM, FDIM);
    pack(nl2_W + (size_t)l * FDIM * FDIM, nl2t[l], FDIM, FDIM);
    pack(nl3_W + (size_t)l * FDIM * FDIM, nl3t[l], FDIM, FDIM);
    pack(el1_W + (size_t)l * FDIM * FDIM, el1t[l], FDIM, FDIM);
  }
  pack(dec0_W, dec0t, XC, XC);
  pack(dec1_W, dec1t, FDIM, FDIM);
  pack(dec2_W, dec2t, FDIM, FDIM);
  pack(dec3_W, dec3t, FDIM, FDIM);

  // Inputs / embeddings.
  k_rbf<<<N_EDGES, 128, 0, stream>>>(efeats, rbf);
  k_node_init<<<CDIV(N_NODES * FDIM, 256), 256, 0, stream>>>(nfeats, node_emb, h, xcat);
  k_edge_init<<<CDIV(N_EDGES * FDIM, 256), 256, 0, stream>>>(nfeats, src, dst, edge_emb, e_bf);

  auto gemm = [&](int act, const __bf16* A, int lda, const __bf16* Bt,
                  const float* bias, const float* resid, float* outF,
                  __bf16* outB, int ldob, int pai, int M, int K) {
    dim3 g(CDIV(M, 32));
    if (act == 0)
      k_gemm_bf16<0><<<g, 256, 0, stream>>>(A, lda, Bt, bias, resid, outF, outB, ldob, prelu_a, pai, M, K);
    else if (act == 1)
      k_gemm_bf16<1><<<g, 256, 0, stream>>>(A, lda, Bt, bias, resid, outF, outB, ldob, prelu_a, pai, M, K);
    else
      k_gemm_bf16<2><<<g, 256, 0, stream>>>(A, lda, Bt, bias, resid, outF, outB, ldob, prelu_a, pai, M, K);
  };

  for (int l = 0; l < NLAYERS; ++l) {
    // d_pre = sp(rbf @ d1_W + b)            [E,512]x[512,128] -> bf16
    gemm(1, rbf, RPAD, d1t[l], d1_b + l * FDIM, nullptr, nullptr, dp_bf, FDIM, 0, N_EDGES, RPAD);
    // d = d_pre @ d2_W + b                  -> f32
    gemm(0, dp_bf, FDIM, d2t[l], d2_b + l * FDIM, nullptr, d_f32, nullptr, FDIM, 0, N_EDGES, FDIM);
    // e2 = e @ eu_W + b                     -> bf16
    gemm(0, e_bf, FDIM, eut[l], eu_b + l * FDIM, nullptr, nullptr, e2_bf, FDIM, 0, N_EDGES, FDIM);
    // hn = h @ nl1_W + b  (h_l = concat slice l, stride 512) -> bf16
    gemm(0, xcat + l * FDIM, XC, nl1t[l], nl1_b + l * FDIM, nullptr, nullptr, hn_bf, FDIM, 0, N_NODES, FDIM);
    // agg = segment_sum(hn[src]*d + e2, dst)
    k_zero_f32<<<CDIV(N_NODES * FDIM, 256), 256, 0, stream>>>(agg, N_NODES * FDIM);
    k_msg_agg<<<CDIV(N_EDGES * 32, 256), 256, 0, stream>>>(hn_bf, d_f32, e2_bf, src, dst, agg);
    k_f32_to_bf16<<<CDIV(N_NODES * FDIM, 256), 256, 0, stream>>>(agg, aggbf, N_NODES * FDIM);
    // t = sp(agg @ nl2_W + b)               -> bf16
    gemm(1, aggbf, FDIM, nl2t[l], nl2_b + l * FDIM, nullptr, nullptr, t_bf, FDIM, 0, N_NODES, FDIM);
    // h = h + (t @ nl3_W + b); also write bf16 h into concat slice l+1
    gemm(0, t_bf, FDIM, nl3t[l], nl3_b + l * FDIM, h, h, xcat + (l + 1) * FDIM, XC, 0, N_NODES, FDIM);
    // e = sp(e2 @ el1_W + b)                -> bf16 (overwrites e)
    gemm(1, e2_bf, FDIM, el1t[l], el1_b + l * FDIM, nullptr, nullptr, e_bf, FDIM, 0, N_EDGES, FDIM);
  }

  // Decoder: prelu chain over the [N,512] concat.
  gemm(2, xcat, XC, dec0t, dec0_b, nullptr, nullptr, decA, FDIM, 0, N_NODES, XC);
  gemm(2, decA, FDIM, dec1t, dec1_b, nullptr, nullptr, decB, FDIM, 1, N_NODES, FDIM);
  gemm(2, decB, FDIM, dec2t, dec2_b, nullptr, nullptr, decA, FDIM, 2, N_NODES, FDIM);
  gemm(2, decA, FDIM, dec3t, dec3_b, nullptr, nullptr, decB, FDIM, 3, N_NODES, FDIM);
  k_dec4<<<CDIV(N_NODES, 256), 256, 0, stream>>>(decB, dec4_W, dec4_b, out);

  (void)in_sizes; (void)n_in; (void)out_size; (void)ws_size;
}
